// LATTE_30408368455823
// MI455X (gfx1250) — matrix operand order repaired and checked by reading, not compiled, not run on hardware
//
#include <hip/hip_runtime.h>
#include <hip/hip_bf16.h>

// ---------------------------------------------------------------------------
// Types for CDNA5 WMMA / TDM
// ---------------------------------------------------------------------------
typedef __attribute__((ext_vector_type(16))) __bf16 v16bf;
typedef __attribute__((ext_vector_type(8)))  float  v8f;
typedef __attribute__((ext_vector_type(4)))  unsigned int v4u;
typedef __attribute__((ext_vector_type(8)))  int v8i_t;
typedef __attribute__((ext_vector_type(4)))  int v4i_t;

#define H_HEADS 8
#define C_DIM   32
#define D_MODEL 256

__device__ __forceinline__ float leaky02(float x) { return x > 0.0f ? x : 0.2f * x; }

// ---------------------------------------------------------------------------
// TDM: 2D fp32 tile (tile_d0 x tile_d1, row stride `stride0` elements) from
// global `gaddr` into LDS offset `lds_off`, compacted row-major.
// D# layout per CDNA5 ISA 8.3/8.4 (group0 128b, group1 256b; groups 2/3 zero,
// legal since tile_dim2 == 0 -> 2D walk only).
// ---------------------------------------------------------------------------
__device__ __forceinline__ void tdm_load_2d_f32(unsigned lds_off, const void* gaddr,
                                                unsigned tensor_d0, unsigned tensor_d1,
                                                unsigned stride0,
                                                unsigned tile_d0, unsigned tile_d1) {
    unsigned long long ga = (unsigned long long)gaddr;
    v4u g0;
    g0[0] = 1u;                                    // count=1, user descriptor
    g0[1] = lds_off;                               // lds_addr
    g0[2] = (unsigned)(ga & 0xffffffffu);          // global_addr[31:0]
    g0[3] = (unsigned)((ga >> 32) & 0x01ffffffu)   // global_addr[56:32]
          | 0x80000000u;                           // type=2 ("image") in [127:126]
    v8i_t g1;
    g1[0] = (int)(2u << 16);                       // wg_mask=0, data_size=4B, no flags
    g1[1] = (int)((tensor_d0 & 0xffffu) << 16);    // tensor_dim0[15:0] @ bits 63:48
    g1[2] = (int)((tensor_d0 >> 16) | ((tensor_d1 & 0xffffu) << 16)); // dim0 hi | dim1 lo
    g1[3] = (int)((tensor_d1 >> 16) | (tile_d0 << 16));               // dim1 hi | tile_dim0
    g1[4] = (int)(tile_d1 & 0xffffu);              // tile_dim1 ; tile_dim2 = 0
    g1[5] = (int)stride0;                          // tensor_dim0_stride[31:0]
    g1[6] = 0;                                     // stride0[47:32]=0 ; dim1_stride lo = 0
    g1[7] = 0;                                     // dim1_stride hi = 0 (unused, 2D)
    v4i_t z4 = {0, 0, 0, 0};
#if __clang_major__ >= 23
    v8i_t z8 = {0, 0, 0, 0, 0, 0, 0, 0};
    __builtin_amdgcn_tensor_load_to_lds(g0, g1, z4, z4, z8, 0);
#else
    __builtin_amdgcn_tensor_load_to_lds(g0, g1, z4, z4, 0);
#endif
}

// ---------------------------------------------------------------------------
// 0) fill kernel (scratch init; harness does not zero between replays)
// ---------------------------------------------------------------------------
__global__ void fill_f32(float* __restrict__ p, long n, float v) {
    long i = (long)blockIdx.x * blockDim.x + threadIdx.x;
    if (i < n) p[i] = v;
}

// ---------------------------------------------------------------------------
// 1) GEMM + bias: Hout[M,256] = X[M,256] @ W[256,256] + b
//    Block: 256 thr = 8 waves, tile 64(M) x 64(N), K-step 32.
//    Tiles staged via Tensor Data Mover (tensor_load_to_lds, TENSORcnt),
//    fp32 in LDS, converted to bf16 at fragment build, then
//    v_wmma_f32_16x16x32_bf16 (2 accumulators / wave, K fully unrolled).
// ---------------------------------------------------------------------------
#define TILE_M 64
#define TILE_N 64
#define TILE_K 32

__global__ __launch_bounds__(256)
void gemm_bias_wmma(const float* __restrict__ X, const float* __restrict__ W,
                    const float* __restrict__ bias, float* __restrict__ Hout,
                    int M) {
    constexpr int K = D_MODEL, N = D_MODEL;
    __shared__ __align__(16) float Af[TILE_M][TILE_K];   // [m][k] fp32, TDM-staged
    __shared__ __align__(16) float Bf[TILE_K][TILE_N];   // [k][n] fp32, TDM-staged

    const int tid  = threadIdx.x;
    const int m0   = blockIdx.x * TILE_M;
    const int n0   = blockIdx.y * TILE_N;

    const int lane = tid & 31;
    const int wv   = tid >> 5;
    const int nt   = wv & 3;                 // N-tile 0..3
    const int mp   = wv >> 2;                // M-pair 0..1 -> M-tiles {2mp, 2mp+1}
    const int l15  = lane & 15;
    const int kh   = lane >> 4;              // 0: K 0-7/16-23 ; 1: K 8-15/24-31

    const unsigned ldsA = (unsigned)(size_t)&Af[0][0];
    const unsigned ldsB = (unsigned)(size_t)&Bf[0][0];

    v8f acc0 = {}, acc1 = {};

#pragma unroll
    for (int kk = 0; kk < K; kk += TILE_K) {
        if (tid == 0) {
            // A tile: rows m0..m0+63, cols kk..kk+31 of X[M,K]
            tdm_load_2d_f32(ldsA, X + (size_t)m0 * K + kk,
                            (unsigned)(K - kk), (unsigned)(M - m0), (unsigned)K,
                            TILE_K, TILE_M);
            // B tile: rows kk..kk+31, cols n0..n0+63 of W[K,N]
            tdm_load_2d_f32(ldsB, W + (size_t)kk * N + n0,
                            (unsigned)(N - n0), (unsigned)(K - kk), (unsigned)N,
                            TILE_N, TILE_K);
        }
        __builtin_amdgcn_s_wait_tensorcnt(0);   // per-wave; non-issuing waves pass
        __syncthreads();

        // B fragment (shared by both accumulators): column n, K = {kh*8..+7, 16+kh*8..+7}
        const int bn = nt * 16 + l15;
        v16bf bfr;
#pragma unroll
        for (int j = 0; j < 16; ++j) bfr[j] = (__bf16)Bf[(j < 8 ? kh * 8 + j : 8 + kh * 8 + j)][bn];

        // A fragments: row m, K = {kh*8..+7, 16+kh*8..+7}
        const int ma0 = (mp * 2 + 0) * 16 + l15;
        const int ma1 = (mp * 2 + 1) * 16 + l15;
        v16bf af0, af1;
#pragma unroll
        for (int j = 0; j < 8; ++j) {
            af0[j]     = (__bf16)Af[ma0][kh * 8 + j];
            af0[8 + j] = (__bf16)Af[ma0][16 + kh * 8 + j];
            af1[j]     = (__bf16)Af[ma1][kh * 8 + j];
            af1[8 + j] = (__bf16)Af[ma1][16 + kh * 8 + j];
        }

        acc0 = __builtin_amdgcn_wmma_f32_16x16x32_bf16(false, af0, false, bfr,
                                                       (short)0, acc0, false, false);
        acc1 = __builtin_amdgcn_wmma_f32_16x16x32_bf16(false, af1, false, bfr,
                                                       (short)0, acc1, false, false);
        __syncthreads();
    }

    // epilogue: D layout lane<16 -> M=r, lane>=16 -> M=8+r (32-bit indexing)
    const int col = n0 + nt * 16 + l15;
    const float bvv = bias[col];
    const int mb0 = m0 + (mp * 2 + 0) * 16 + kh * 8;
    const int mb1 = m0 + (mp * 2 + 1) * 16 + kh * 8;
#pragma unroll
    for (int r = 0; r < 8; ++r) {
        if (mb0 + r < M) Hout[(mb0 + r) * N + col] = acc0[r] + bvv;
        if (mb1 + r < M) Hout[(mb1 + r) * N + col] = acc1[r] + bvv;
    }
}

// ---------------------------------------------------------------------------
// 2) per-(node,head) attention scalar:  out[n,h] = sum_c h[n,h,c]*vec[h,c]
// ---------------------------------------------------------------------------
__global__ void node_head_dot(const float* __restrict__ h, const float* __restrict__ vec,
                              float* __restrict__ out, int n_nodes) {
    int i = blockIdx.x * blockDim.x + threadIdx.x;
    if (i >= n_nodes * H_HEADS) return;
    int n = i >> 3, hd = i & 7;
    const float* hp = h + (size_t)n * D_MODEL + hd * C_DIM;
    const float* vp = vec + hd * C_DIM;
    float s = 0.0f;
#pragma unroll
    for (int c = 0; c < C_DIM; ++c) s += hp[c] * vp[c];
    out[i] = s;
}

// ---------------------------------------------------------------------------
// 3) edge passes (GAT segment-softmax, recompute logits each pass)
// ---------------------------------------------------------------------------
__device__ __forceinline__ void atomicMaxF(float* addr, float val) {
    int* ia = (int*)addr;
    int old = __float_as_int(*addr);
    while (__int_as_float(old) < val) {
        int assumed = old;
        old = atomicCAS(ia, assumed, __float_as_int(val));
        if (old == assumed) break;
    }
}

__global__ void edge_max_kernel(const int* __restrict__ edge, int E,
                                const float* __restrict__ al, const float* __restrict__ ar,
                                float* __restrict__ m) {
    int i = blockIdx.x * blockDim.x + threadIdx.x;
    if (i >= E * H_HEADS) return;
    int e = i >> 3, hd = i & 7;
    int src = edge[e], dst = edge[E + e];
    float v = leaky02(al[src * H_HEADS + hd] + ar[dst * H_HEADS + hd]);
    atomicMaxF(&m[dst * H_HEADS + hd], v);
}

__global__ void edge_expsum_kernel(const int* __restrict__ edge, int E,
                                   const float* __restrict__ al, const float* __restrict__ ar,
                                   const float* __restrict__ m, float* __restrict__ denom) {
    int i = blockIdx.x * blockDim.x + threadIdx.x;
    if (i >= E * H_HEADS) return;
    int e = i >> 3, hd = i & 7;
    int src = edge[e], dst = edge[E + e];
    float v  = leaky02(al[src * H_HEADS + hd] + ar[dst * H_HEADS + hd]);
    float ex = __expf(v - m[dst * H_HEADS + hd]);
    atomicAdd(&denom[dst * H_HEADS + hd], ex);
}

// one block (256 thr) per edge: thread t handles (head=t>>5, c=t&31)
__global__ __launch_bounds__(256)
void edge_scatter_kernel(const int* __restrict__ edge, int E,
                         const float* __restrict__ al, const float* __restrict__ ar,
                         const float* __restrict__ m, const float* __restrict__ denom,
                         const float* __restrict__ hsrc, float* __restrict__ rel) {
    int e = blockIdx.x;
    if (e >= E) return;
    int t = threadIdx.x;
    int hd = t >> 5;
    int src = edge[e], dst = edge[E + e];
    __builtin_prefetch(hsrc + (size_t)src * D_MODEL + t, 0, 1);
    float v     = leaky02(al[src * H_HEADS + hd] + ar[dst * H_HEADS + hd]);
    float alpha = __expf(v - m[dst * H_HEADS + hd]) / denom[dst * H_HEADS + hd];
    atomicAdd(&rel[(size_t)dst * D_MODEL + t], hsrc[(size_t)src * D_MODEL + t] * alpha);
}

// ---------------------------------------------------------------------------
// 4) relation-level beta softmax + ReLU  (A: 3 relations, B: 2 relations)
// ---------------------------------------------------------------------------
__global__ void beta_A_kernel(const float* __restrict__ hA,
                              const float* __restrict__ rel_ba, const float* __restrict__ rel_aa,
                              const float* __restrict__ rel_l, const float* __restrict__ rel_r,
                              float* __restrict__ out, int nA) {
    int i = blockIdx.x * blockDim.x + threadIdx.x;
    if (i >= nA * H_HEADS) return;
    int n = i >> 3, hd = i & 7;
    const float* h  = hA     + (size_t)n * D_MODEL + hd * C_DIM;
    const float* e0 = rel_ba + (size_t)n * D_MODEL + hd * C_DIM;
    const float* e1 = rel_aa + (size_t)n * D_MODEL + hd * C_DIM;
    const float* rl = rel_l + hd * C_DIM;
    const float* rr = rel_r + hd * C_DIM;
    float bl = 0, b0 = 0, b1 = 0, b2 = 0;
#pragma unroll
    for (int c = 0; c < C_DIM; ++c) {
        bl += h[c] * rl[c];
        b0 += e0[c] * rr[c];
        b1 += e1[c] * rr[c];
        b2 += h[c] * rr[c];
    }
    float s0 = leaky02(bl + b0), s1 = leaky02(bl + b1), s2 = leaky02(bl + b2);
    float mx = fmaxf(s0, fmaxf(s1, s2));
    float x0 = __expf(s0 - mx), x1 = __expf(s1 - mx), x2 = __expf(s2 - mx);
    float inv = 1.0f / (x0 + x1 + x2);
    x0 *= inv; x1 *= inv; x2 *= inv;
    float* op = out + (size_t)n * D_MODEL + hd * C_DIM;
#pragma unroll
    for (int c = 0; c < C_DIM; ++c) {
        float v = x0 * e0[c] + x1 * e1[c] + x2 * h[c];
        op[c] = v > 0.0f ? v : 0.0f;
    }
}

__global__ void beta_B_kernel(const float* __restrict__ hB, const float* __restrict__ rel_ab,
                              const float* __restrict__ rel_l, const float* __restrict__ rel_r,
                              float* __restrict__ out, int nB, int rowOff) {
    int i = blockIdx.x * blockDim.x + threadIdx.x;
    if (i >= nB * H_HEADS) return;
    int n = i >> 3, hd = i & 7;
    const float* h  = hB     + (size_t)n * D_MODEL + hd * C_DIM;
    const float* e0 = rel_ab + (size_t)n * D_MODEL + hd * C_DIM;
    const float* rl = rel_l + hd * C_DIM;
    const float* rr = rel_r + hd * C_DIM;
    float bl = 0, b0 = 0, b1 = 0;
#pragma unroll
    for (int c = 0; c < C_DIM; ++c) {
        bl += h[c] * rl[c];
        b0 += e0[c] * rr[c];
        b1 += h[c] * rr[c];
    }
    float s0 = leaky02(bl + b0), s1 = leaky02(bl + b1);
    float mx = fmaxf(s0, s1);
    float x0 = __expf(s0 - mx), x1 = __expf(s1 - mx);
    float inv = 1.0f / (x0 + x1);
    x0 *= inv; x1 *= inv;
    float* op = out + (size_t)(rowOff + n) * D_MODEL + hd * C_DIM;
#pragma unroll
    for (int c = 0; c < C_DIM; ++c) {
        float v = x0 * e0[c] + x1 * h[c];
        op[c] = v > 0.0f ? v : 0.0f;
    }
}

// ---------------------------------------------------------------------------
// launcher
// ---------------------------------------------------------------------------
extern "C" void kernel_launch(void* const* d_in, const int* in_sizes, int n_in,
                              void* d_out, int out_size, void* d_ws, size_t ws_size,
                              hipStream_t stream) {
    const float* x_A     = (const float*)d_in[0];
    const float* x_B     = (const float*)d_in[1];
    const int*   edge_ab = (const int*)  d_in[2];
    const int*   edge_ba = (const int*)  d_in[3];
    const int*   edge_aa = (const int*)  d_in[4];
    const float* W_A     = (const float*)d_in[5];
    const float* b_A     = (const float*)d_in[6];
    const float* W_B     = (const float*)d_in[7];
    const float* b_B     = (const float*)d_in[8];
    const float* attn_l  = (const float*)d_in[9];   // [3,8,32]
    const float* attn_r  = (const float*)d_in[10];  // [3,8,32]
    const float* rel_l_A = (const float*)d_in[11];
    const float* rel_r_A = (const float*)d_in[12];
    const float* rel_l_B = (const float*)d_in[13];
    const float* rel_r_B = (const float*)d_in[14];
    float* out = (float*)d_out;

    const int NA = in_sizes[0] / D_MODEL;
    const int NB = in_sizes[1] / D_MODEL;
    const int E  = in_sizes[2] / 2;

    // ---- workspace carve-up (floats) ----
    float* ws = (float*)d_ws;
    size_t off = 0;
    float* hA  = ws + off; off += (size_t)NA * D_MODEL;
    float* hB  = ws + off; off += (size_t)NB * D_MODEL;
    float* al0 = ws + off; off += (size_t)NA * H_HEADS;   // A src of ab
    float* ar0 = ws + off; off += (size_t)NB * H_HEADS;   // B dst of ab
    float* al1 = ws + off; off += (size_t)NB * H_HEADS;   // B src of ba
    float* ar1 = ws + off; off += (size_t)NA * H_HEADS;   // A dst of ba
    float* al2 = ws + off; off += (size_t)NA * H_HEADS;   // A src of aa
    float* ar2 = ws + off; off += (size_t)NA * H_HEADS;   // A dst of aa
    // -inf-init region (segment maxima)
    float* m_ba = ws + off; off += (size_t)NA * H_HEADS;
    float* m_aa = ws + off; off += (size_t)NA * H_HEADS;
    float* m_ab = ws + off; off += (size_t)NB * H_HEADS;
    const size_t m_region = (size_t)(2 * NA + NB) * H_HEADS;
    // zero-init region (denoms + relation accumulators), contiguous
    float* zr_start = ws + off;
    float* d_ba = ws + off; off += (size_t)NA * H_HEADS;
    float* d_aa = ws + off; off += (size_t)NA * H_HEADS;
    float* d_ab = ws + off; off += (size_t)NB * H_HEADS;
    float* rel_ba = ws + off; off += (size_t)NA * D_MODEL;
    float* rel_aa = ws + off; off += (size_t)NA * D_MODEL;
    float* rel_ab = ws + off; off += (size_t)NB * D_MODEL;
    const size_t z_region = (size_t)((ws + off) - zr_start);
    (void)ws_size;

    const int BLK = 256;
    auto cdiv = [](long a, long b) { return (int)((a + b - 1) / b); };

    // ---- init scratch ----
    fill_f32<<<cdiv(m_region, BLK), BLK, 0, stream>>>(m_ba, (long)m_region, -3.0e38f);
    fill_f32<<<cdiv(z_region, BLK), BLK, 0, stream>>>(zr_start, (long)z_region, 0.0f);

    // ---- projections (TDM-staged WMMA) ----
    dim3 ggridA(cdiv(NA, TILE_M), D_MODEL / TILE_N);
    gemm_bias_wmma<<<ggridA, BLK, 0, stream>>>(x_A, W_A, b_A, hA, NA);
    dim3 ggridB(cdiv(NB, TILE_M), D_MODEL / TILE_N);
    gemm_bias_wmma<<<ggridB, BLK, 0, stream>>>(x_B, W_B, b_B, hB, NB);

    // ---- per-node attention scalars ----
    const int ga = cdiv((long)NA * H_HEADS, BLK), gb = cdiv((long)NB * H_HEADS, BLK);
    node_head_dot<<<ga, BLK, 0, stream>>>(hA, attn_l + 0 * D_MODEL, al0, NA); // mp0: A->B
    node_head_dot<<<gb, BLK, 0, stream>>>(hB, attn_r + 0 * D_MODEL, ar0, NB);
    node_head_dot<<<gb, BLK, 0, stream>>>(hB, attn_l + 1 * D_MODEL, al1, NB); // mp1: B->A
    node_head_dot<<<ga, BLK, 0, stream>>>(hA, attn_r + 1 * D_MODEL, ar1, NA);
    node_head_dot<<<ga, BLK, 0, stream>>>(hA, attn_l + 2 * D_MODEL, al2, NA); // mp2: A->A
    node_head_dot<<<ga, BLK, 0, stream>>>(hA, attn_r + 2 * D_MODEL, ar2, NA);

    // ---- GAT segment softmax + scatter, per relation ----
    const int ge = cdiv((long)E * H_HEADS, BLK);
    // relation ba: src in B, dst in A
    edge_max_kernel   <<<ge, BLK, 0, stream>>>(edge_ba, E, al1, ar1, m_ba);
    edge_expsum_kernel<<<ge, BLK, 0, stream>>>(edge_ba, E, al1, ar1, m_ba, d_ba);
    edge_scatter_kernel<<<E, BLK, 0, stream>>>(edge_ba, E, al1, ar1, m_ba, d_ba, hB, rel_ba);
    // relation aa: src in A, dst in A
    edge_max_kernel   <<<ge, BLK, 0, stream>>>(edge_aa, E, al2, ar2, m_aa);
    edge_expsum_kernel<<<ge, BLK, 0, stream>>>(edge_aa, E, al2, ar2, m_aa, d_aa);
    edge_scatter_kernel<<<E, BLK, 0, stream>>>(edge_aa, E, al2, ar2, m_aa, d_aa, hA, rel_aa);
    // relation ab: src in A, dst in B
    edge_max_kernel   <<<ge, BLK, 0, stream>>>(edge_ab, E, al0, ar0, m_ab);
    edge_expsum_kernel<<<ge, BLK, 0, stream>>>(edge_ab, E, al0, ar0, m_ab, d_ab);
    edge_scatter_kernel<<<E, BLK, 0, stream>>>(edge_ab, E, al0, ar0, m_ab, d_ab, hA, rel_ab);

    // ---- relation-level beta combine + ReLU -> d_out ----
    beta_A_kernel<<<ga, BLK, 0, stream>>>(hA, rel_ba, rel_aa, rel_l_A, rel_r_A, out, NA);
    beta_B_kernel<<<gb, BLK, 0, stream>>>(hB, rel_ab, rel_l_B, rel_r_B, out, NB, NA);
    (void)n_in; (void)out_size;
}
